// SimpleGCN_41532333752971
// MI455X (gfx1250) — matrix operand order repaired and checked by reading, not compiled, not run on hardware
//
#include <hip/hip_runtime.h>

typedef __attribute__((ext_vector_type(16))) _Float16 v16h;
typedef __attribute__((ext_vector_type(4)))  _Float16 v4h;
typedef __attribute__((ext_vector_type(2)))  _Float16 v2h;
typedef __attribute__((ext_vector_type(8)))  float    v8f;

#define FEAT1 128
#define FEAT2 64
#define KDIM  128

// ---------------- degree / normalization ----------------

__global__ void fill_ones_kernel(float* __restrict__ p, int n) {
  int i = blockIdx.x * blockDim.x + threadIdx.x;
  if (i < n) p[i] = 1.0f;   // self-loop contributes 1 to every node's degree
}

__global__ void deg_count_kernel(const int* __restrict__ dst, float* __restrict__ deg, int E) {
  int e = blockIdx.x * blockDim.x + threadIdx.x;
  if (e < E) atomicAdd(&deg[dst[e]], 1.0f);
}

__global__ void deg_to_dinv_kernel(float* __restrict__ deg, int n) {
  int i = blockIdx.x * blockDim.x + threadIdx.x;
  if (i < n) {
    float d = deg[i];
    deg[i] = (d > 0.0f) ? rsqrtf(d) : 0.0f;
  }
}

// ---------------- B-fragment pre-swizzle ----------------
// Pack W[K=128, N] (f32, row-major) into this file's contiguous-k per-lane WMMA B-fragment
// layout: flat half index = (((tn*4 + kb32)*32 + lane)*16 + e)
//   k = kb32*32 + e + (lane/16)*16 ; n = tn*16 + (lane%16)
// In the GEMM each lane then reads one contiguous 32B v16h per (tn, K-step).

__global__ void pack_B_kernel(const float* __restrict__ W, _Float16* __restrict__ Bp, int N) {
  int idx = blockIdx.x * blockDim.x + threadIdx.x;
  int total = (N >> 4) * (KDIM / 32) * 32 * 16;
  if (idx >= total) return;
  int e    = idx & 15;
  int lane = (idx >> 4) & 31;
  int kb32 = (idx >> 9) & 3;
  int tn   = idx >> 11;
  int k = kb32 * 32 + e + ((lane >> 4) << 4);
  int n = (tn << 4) + (lane & 15);
  Bp[idx] = (_Float16)W[k * N + n];
}

// ---------------- WMMA GEMM strip kernel + fused epilogue ----------------
// One wave owns a 16-row strip across ALL N columns: the A fragment is loaded
// once per K-step and reused by TILESN WMMAs (A traffic /TILESN).
//   Ch[M,N] (f16) = act(A[M,128]) @ W      act = ReLU if RELU (layer-2 input)
//   Of[M,N] (f32) = dinv[r]^2 * Ch + bias  (self-loop term of GCN aggregation)

template<int TILESN, bool RELU>
__global__ void gemm_wmma_strip_kernel(const float* __restrict__ A,
                                       const v16h* __restrict__ Bpack,
                                       const float* __restrict__ dinv,
                                       const float* __restrict__ bias,
                                       _Float16* __restrict__ Ch,
                                       float* __restrict__ Of,
                                       int M) {
  const int N = TILESN * 16;
  int waveId = (int)((blockIdx.x * blockDim.x + threadIdx.x) >> 5);
  int lane   = threadIdx.x & 31;
  int tilesM = (M + 15) >> 4;
  if (waveId >= tilesM) return;

  int m0  = waveId << 4;
  int hi  = lane >> 4;   // 0: lanes 0-15, 1: lanes 16-31
  int l15 = lane & 15;

  int arowIdx = m0 + l15;
  if (arowIdx >= M) arowIdx = M - 1;            // clamp (M is a multiple of 16 here)
  const float* arow = A + (size_t)arowIdx * KDIM;

  v8f acc[TILESN];
#pragma unroll
  for (int t = 0; t < TILESN; ++t) acc[t] = (v8f){};

#pragma unroll
  for (int kb32 = 0; kb32 < KDIM / 32; ++kb32) {
    // A fragment: contiguous k order, matching the B pack (lines 46/90):
    // lane L holds row M=L%16; element e -> K = e + (L/16)*16
    v16h a;
#pragma unroll
    for (int e = 0; e < 16; ++e) {
      int k = kb32 * 32 + e + (hi << 4);
      float av = arow[k];
      if (RELU) av = fmaxf(av, 0.0f);           // fused ReLU on layer-2 input
      a[e] = (_Float16)av;
    }
    // TILESN B fragments, each one contiguous 32B pre-swizzled read
#pragma unroll
    for (int t = 0; t < TILESN; ++t) {
      v16h b = Bpack[(size_t)(t * (KDIM / 32) + kb32) * 32 + lane];
      acc[t] = __builtin_amdgcn_wmma_f32_16x16x32_f16(
          /*neg_a=*/false, a, /*neg_b=*/false, b,
          /*c_mod=*/(short)0, acc[t], /*reuse_a=*/false, /*reuse_b=*/false);
    }
  }

  // C/D layout: lane L, VGPR v -> row m0 + v + (L/16)*8, col tn*16 + L%16
  float dv[8];
#pragma unroll
  for (int v = 0; v < 8; ++v) {
    int r = m0 + v + (hi << 3);
    dv[v] = dinv[r < M ? r : (M - 1)];
  }
#pragma unroll
  for (int t = 0; t < TILESN; ++t) {
    int   col = t * 16 + l15;
    float bc  = bias[col];
#pragma unroll
    for (int v = 0; v < 8; ++v) {
      int r = m0 + v + (hi << 3);
      if (r < M) {
        size_t o = (size_t)r * N + col;
        Ch[o] = (_Float16)acc[t][v];
        Of[o] = dv[v] * dv[v] * acc[t][v] + bc;   // self-loop message + bias
      }
    }
  }
}

// ---------------- edge aggregation (one wave32 per edge) ----------------
// out[dst] += dinv[src]*dinv[dst] * h[src]; lane owns F/32 contiguous features.

__global__ void agg_edge128_kernel(const int* __restrict__ src,
                                   const int* __restrict__ dst,
                                   const _Float16* __restrict__ h,
                                   const float* __restrict__ dinv,
                                   float* __restrict__ out, int E) {
  int wave = (int)((blockIdx.x * blockDim.x + threadIdx.x) >> 5);
  int lane = threadIdx.x & 31;
  if (wave >= E) return;
  int s = src[wave];
  int d = dst[wave];
  float norm = dinv[s] * dinv[d];
  const v4h hv = *(const v4h*)(h + (size_t)s * 128 + lane * 4);  // 8B coalesced load
  float* o = out + (size_t)d * 128 + lane * 4;
#pragma unroll
  for (int j = 0; j < 4; ++j)
    atomicAdd(&o[j], norm * (float)hv[j]);   // global_atomic_add_f32 (L2-resident)
}

__global__ void agg_edge64_kernel(const int* __restrict__ src,
                                  const int* __restrict__ dst,
                                  const _Float16* __restrict__ h,
                                  const float* __restrict__ dinv,
                                  float* __restrict__ out, int E) {
  int wave = (int)((blockIdx.x * blockDim.x + threadIdx.x) >> 5);
  int lane = threadIdx.x & 31;
  if (wave >= E) return;
  int s = src[wave];
  int d = dst[wave];
  float norm = dinv[s] * dinv[d];
  const v2h hv = *(const v2h*)(h + (size_t)s * 64 + lane * 2);   // 4B coalesced load
  float* o = out + (size_t)d * 64 + lane * 2;
#pragma unroll
  for (int j = 0; j < 2; ++j)
    atomicAdd(&o[j], norm * (float)hv[j]);
}

// ---------------- launch ----------------

extern "C" void kernel_launch(void* const* d_in, const int* in_sizes, int n_in,
                              void* d_out, int out_size, void* d_ws, size_t ws_size,
                              hipStream_t stream) {
  const float* x   = (const float*)d_in[0];
  const int*  eidx = (const int*)d_in[1];
  const float* W1  = (const float*)d_in[2];
  const float* b1  = (const float*)d_in[3];
  const float* W2  = (const float*)d_in[4];
  const float* b2  = (const float*)d_in[5];

  const int N = in_sizes[0] / FEAT1;   // 100000 nodes
  const int E = in_sizes[1] / 2;       // 1600000 edges
  const int* src = eidx;
  const int* dst = eidx + E;

  // workspace carve-out (256B aligned)
  char* ws = (char*)d_ws;
  size_t off = 0;
  auto carve = [&](size_t bytes) -> void* {
    void* p = (void*)(ws + off);
    off = (off + bytes + 255) & ~(size_t)255;
    return p;
  };
  float*    dinv = (float*)carve((size_t)N * sizeof(float));
  _Float16* B1p  = (_Float16*)carve((size_t)(FEAT1 / 16) * 4 * 32 * 16 * sizeof(_Float16));
  _Float16* B2p  = (_Float16*)carve((size_t)(FEAT2 / 16) * 4 * 32 * 16 * sizeof(_Float16));
  _Float16* h1   = (_Float16*)carve((size_t)N * FEAT1 * sizeof(_Float16));
  float*    out1 = (float*)carve((size_t)N * FEAT1 * sizeof(float));
  _Float16* h2   = (_Float16*)carve((size_t)N * FEAT2 * sizeof(_Float16));
  float*    out2 = (float*)d_out;

  const int B  = 256;                 // 8 wave32s per block
  const int WB = B / 32;              // waves per block

  // normalization coefficients
  fill_ones_kernel<<<(N + B - 1) / B, B, 0, stream>>>(dinv, N);
  deg_count_kernel<<<(E + B - 1) / B, B, 0, stream>>>(dst, dinv, E);
  deg_to_dinv_kernel<<<(N + B - 1) / B, B, 0, stream>>>(dinv, N);

  // pre-swizzle weights into WMMA B-fragment layout
  {
    int t1 = (FEAT1 / 16) * 4 * 32 * 16;
    int t2 = (FEAT2 / 16) * 4 * 32 * 16;
    pack_B_kernel<<<(t1 + B - 1) / B, B, 0, stream>>>(W1, B1p, FEAT1);
    pack_B_kernel<<<(t2 + B - 1) / B, B, 0, stream>>>(W2, B2p, FEAT2);
  }

  const int stripWaves = (N + 15) / 16;             // one wave per 16-row strip
  const int stripGrid  = (stripWaves + WB - 1) / WB;

  // ---- layer 1: GEMM (+ self-loop/bias epilogue), edge scatter ----
  gemm_wmma_strip_kernel<FEAT1 / 16, false><<<stripGrid, B, 0, stream>>>(
      x, (const v16h*)B1p, dinv, b1, h1, out1, N);
  agg_edge128_kernel<<<(E + WB - 1) / WB, B, 0, stream>>>(src, dst, h1, dinv, out1, E);

  // ---- layer 2: GEMM with fused ReLU on input, edge scatter to d_out ----
  gemm_wmma_strip_kernel<FEAT2 / 16, true><<<stripGrid, B, 0, stream>>>(
      out1, (const v16h*)B2p, dinv, b2, h2, out2, N);
  agg_edge64_kernel<<<(E + WB - 1) / WB, B, 0, stream>>>(src, dst, h2, dinv, out2, E);
}
